// CHGNetRL_61323543052774
// MI455X (gfx1250) — matrix operand order repaired and checked by reading, not compiled, not run on hardware
//
#include <hip/hip_runtime.h>
#include <stdint.h>

#define F 64
#define N_ATOMS 10000
#define N_UB 100000
#define N_DB 200000
#define N_ANG 100000
#define N_GRAPHS 32
#define N_RBF 31
#define N_ABASIS 21
#define N_CONV 4
#define N_ACTIONS 88

typedef __attribute__((ext_vector_type(16))) __bf16 v16bf;
typedef __attribute__((ext_vector_type(8)))  float  v8f;

union BF16Frag { uint32_t u[8]; uint4 q[2]; v16bf v; };

__device__ __forceinline__ uint16_t f32_to_bf16(float f) {
  uint32_t u = __builtin_bit_cast(uint32_t, f);
  u += 0x7FFFu + ((u >> 16) & 1u);          // round to nearest even
  return (uint16_t)(u >> 16);
}
__device__ __forceinline__ float silu_f(float x) { return x / (1.0f + __expf(-x)); }
__device__ __forceinline__ float sigm_f(float x) { return 1.0f / (1.0f + __expf(-x)); }

// -------------------------------------------------------------------------
// Pack f32 weights [Ktot,64] -> bf16 WMMA B-fragments, fragment-ordered so each
// lane's 8 dwords are contiguous (2x b128 loads):
//   P[((kc*4 + nt)*32 + lane)*8 + v] = pack(W[k0][n], W[k0+1][n])
//   k0 = kc*32 + (lane>>4)*16 + 2v, n = nt*16 + (lane&15)
// Works on concatenated layer stacks (layer K is a multiple of 32).
// -------------------------------------------------------------------------
__global__ void pack_frag_kernel(const float* __restrict__ W,
                                 uint32_t* __restrict__ P, int Ktot) {
  int t = blockIdx.x * blockDim.x + threadIdx.x;
  int total = (Ktot / 32) * 1024;         // dwords
  if (t >= total) return;
  int v    = t & 7;
  int lane = (t >> 3) & 31;
  int nt   = (t >> 8) & 3;
  int kc   = t >> 10;
  int k0 = kc * 32 + (lane >> 4) * 16 + 2 * v;
  int n  = nt * 16 + (lane & 15);
  uint16_t lo = f32_to_bf16(W[k0 * F + n]);
  uint16_t hi = f32_to_bf16(W[(k0 + 1) * F + n]);
  P[t] = ((uint32_t)hi << 16) | (uint32_t)lo;
}

// -------------------------------------------------------------------------
// Gated message kernel: rows of X = concat of NSEG gathered 64-wide features.
// msg = silu(X@Wc+bc) * sigmoid(X@Wg+bg) [* wscale[scaleIdx[row]]]
// Scatter-add (segment sum) via atomics, or direct row add.
// One WG = 128 threads = 4 waves; 64 rows per WG; WMMA bf16 16x16x32.
// -------------------------------------------------------------------------
template <int NSEG, bool HAS_SCALE, bool SCATTER>
__global__ void __launch_bounds__(128)
gated_msg_kernel(int nRows,
                 const float* __restrict__ s0, const int* __restrict__ i0,
                 const float* __restrict__ s1, const int* __restrict__ i1,
                 const float* __restrict__ s2, const int* __restrict__ i2,
                 const float* __restrict__ s3, const int* __restrict__ i3,
                 const uint32_t* __restrict__ WcP, const float* __restrict__ bc,
                 const uint32_t* __restrict__ WgP, const float* __restrict__ bg,
                 const float* __restrict__ wscale, const int* __restrict__ scaleIdx,
                 float* __restrict__ outp, const int* __restrict__ scatIdx) {
  constexpr int K  = NSEG * F;   // 192 or 256
  constexpr int KC = K / 32;     // WMMA K-chunks

  __shared__ uint16_t sX[64 * K];   // bf16 X tile, row-major
  __shared__ int sIdx[4][64];       // gather index per (seg,row)
  __shared__ int sScat[64];
  __shared__ int sScl[64];

  const int tid = threadIdx.x;
  const int rowBase = blockIdx.x * 64;

  // Stage per-row indices.
  if (tid < 64) {
    int r = rowBase + tid;
    bool ok = r < nRows;
    if (SCATTER)   sScat[tid] = ok ? scatIdx[r] : 0;
    if (HAS_SCALE) sScl[tid]  = ok ? scaleIdx[r] : 0;
    #pragma unroll
    for (int s = 0; s < NSEG; s++) {
      const int* ip = (s == 0) ? i0 : (s == 1) ? i1 : (s == 2) ? i2 : i3;
      sIdx[s][tid] = ok ? (ip ? ip[r] : r) : -1;
    }
  }
  __syncthreads();

  // Vectorized gather: one float4 (b128) per iteration -> 2 packed-bf16 dwords.
  constexpr int NQ = 64 * K / 4;    // float4 count in the tile
  for (int q = tid; q < NQ; q += 128) {
    int f4 = q & 15;                // float4 within 64-wide segment
    int rs = q >> 4;                // row-segment
    int r  = rs / NSEG;
    int s  = rs - r * NSEG;
    int gi = sIdx[s][r];
    float4 val = {0.0f, 0.0f, 0.0f, 0.0f};
    if (gi >= 0) {
      const float* sp = (s == 0) ? s0 : (s == 1) ? s1 : (s == 2) ? s2 : s3;
      val = ((const float4*)(sp + (size_t)gi * F))[f4];
    }
    uint32_t d0 = ((uint32_t)f32_to_bf16(val.y) << 16) | (uint32_t)f32_to_bf16(val.x);
    uint32_t d1 = ((uint32_t)f32_to_bf16(val.w) << 16) | (uint32_t)f32_to_bf16(val.z);
    uint2 u2; u2.x = d0; u2.y = d1;
    *(uint2*)&sX[r * K + s * F + f4 * 4] = u2;    // 8B-aligned (K, F even; f4*4*2B)
  }
  __syncthreads();

  const int lane = tid & 31;
  const int wave = tid >> 5;
  const int lm = lane & 15;   // lane within half-wave
  const int lg = lane >> 4;   // half-wave group
  const int aRow = wave * 16 + lm;   // X row within WG tile for A frag
  const uint32_t* xp = (const uint32_t*)sX;
  const uint4* WcV = (const uint4*)WcP;
  const uint4* WgV = (const uint4*)WgP;

  v8f accC[4] = {};
  v8f accG[4] = {};

  #pragma unroll
  for (int kc = 0; kc < KC; kc++) {
    const int kbase = kc * 32;
    // A fragment 16x32 bf16: half-wave 0 holds K 0-15, half-wave 1 holds
    // K 16-31; each VGPR = 2 consecutive bf16 (dwords from LDS -> b128).
    BF16Frag a;
    const int abase = aRow * (K / 2) + (kbase >> 1) + lg * 8;
    #pragma unroll
    for (int v = 0; v < 4; v++) a.u[v]     = xp[abase + v];
    #pragma unroll
    for (int v = 0; v < 4; v++) a.u[4 + v] = xp[abase + 4 + v];

    #pragma unroll
    for (int nt = 0; nt < 4; nt++) {
      // Fragment-ordered B: lane's 8 dwords contiguous -> two b128 loads.
      const int fbase = (((kc * 4 + nt) * 32 + lane) * 8) >> 2;  // uint4 units
      BF16Frag b;
      b.q[0] = WcV[fbase];
      b.q[1] = WcV[fbase + 1];
      accC[nt] = __builtin_amdgcn_wmma_f32_16x16x32_bf16(
          false, a.v, false, b.v, (short)0, accC[nt], false, false);
      b.q[0] = WgV[fbase];
      b.q[1] = WgV[fbase + 1];
      accG[nt] = __builtin_amdgcn_wmma_f32_16x16x32_bf16(
          false, a.v, false, b.v, (short)0, accG[nt], false, false);
    }
  }

  // Epilogue: C/D layout — lanes 0-15: M=v, N=lane; lanes 16-31: M=v+8, N=lane-16.
  #pragma unroll
  for (int nt = 0; nt < 4; nt++) {
    #pragma unroll
    for (int v = 0; v < 8; v++) {
      int M = v + lg * 8;
      int rloc = wave * 16 + M;
      int row = rowBase + rloc;
      if (row >= nRows) continue;
      int N = nt * 16 + lm;
      float cV = accC[nt][v] + bc[N];
      float gV = accG[nt][v] + bg[N];
      float m = silu_f(cV) * sigm_f(gV);
      if (HAS_SCALE) m *= wscale[(size_t)sScl[rloc] * F + N];
      if (SCATTER) atomicAdd(outp + (size_t)sScat[rloc] * F + N, m);
      else         outp[(size_t)row * F + N] += m;
    }
  }
}

// -------------------------------------------------------------------------
// Small dense: out[r][f] = b[f] + sum_k bases[r*K+k] * W[k*64+f]
// -------------------------------------------------------------------------
__global__ void linear_small_kernel(const float* __restrict__ B,
                                    const float* __restrict__ W,
                                    const float* __restrict__ bias,
                                    float* __restrict__ out, int rows, int K) {
  int t = blockIdx.x * blockDim.x + threadIdx.x;
  if (t >= rows * F) return;
  int r = t >> 6, f = t & 63;
  float acc = bias[f];
  const float* br = B + (size_t)r * K;
  for (int k = 0; k < K; k++) acc = fmaf(br[k], W[k * F + f], acc);
  out[t] = acc;
}

__global__ void atom_embed_kernel(const int* __restrict__ z,
                                  const float* __restrict__ emb,
                                  float* __restrict__ out, int n) {
  int t = blockIdx.x * blockDim.x + threadIdx.x;
  if (t >= n * F) return;
  int a = t >> 6, f = t & 63;
  out[t] = emb[z[a] * F + f];
}

// One wave per row (wave32); lane handles elements [lane] and [lane+32].
__global__ void __launch_bounds__(128)
layernorm_kernel(float* __restrict__ x, const float* __restrict__ g,
                 const float* __restrict__ b, int rows) {
  int row = blockIdx.x * 4 + (threadIdx.x >> 5);
  int lane = threadIdx.x & 31;
  if (row >= rows) return;
  float* rp = x + (size_t)row * F;
  float e0 = rp[lane], e1 = rp[lane + 32];
  float s = e0 + e1, sq = e0 * e0 + e1 * e1;
  #pragma unroll
  for (int off = 16; off > 0; off >>= 1) {
    s  += __shfl_xor(s, off, 32);
    sq += __shfl_xor(sq, off, 32);
  }
  float mean = s * (1.0f / F);
  float var = sq * (1.0f / F) - mean * mean;
  float inv = rsqrtf(var + 1e-5f);
  rp[lane]      = (e0 - mean) * inv * g[lane]      + b[lane];
  rp[lane + 32] = (e1 - mean) * inv * g[lane + 32] + b[lane + 32];
}

__global__ void pool_zero_kernel(float* __restrict__ sums, float* __restrict__ counts) {
  int t = blockIdx.x * blockDim.x + threadIdx.x;
  if (t < N_GRAPHS * F) sums[t] = 0.0f;
  if (t < N_GRAPHS) counts[t] = 0.0f;
}

__global__ void pool_accum_kernel(const float* __restrict__ atom,
                                  const int* __restrict__ owners,
                                  float* __restrict__ sums,
                                  float* __restrict__ counts, int n) {
  int t = blockIdx.x * blockDim.x + threadIdx.x;
  if (t >= n * F) return;
  int a = t >> 6, f = t & 63;
  int o = owners[a];
  atomicAdd(sums + o * F + f, atom[t]);
  if (f == 0) atomicAdd(counts + o, 1.0f);
}

// One block per graph: mean-pool finish + 3-layer silu MLP + output proj.
__global__ void __launch_bounds__(96)
head_kernel(const float* __restrict__ sums, const float* __restrict__ counts,
            const float* __restrict__ W, const float* __restrict__ b,
            const float* __restrict__ Wout, const float* __restrict__ bout,
            float* __restrict__ out) {
  __shared__ float h[F], h2[F];
  int g = blockIdx.x, t = threadIdx.x;
  if (t < F) h[t] = sums[g * F + t] / fmaxf(counts[g], 1.0f);
  __syncthreads();
  for (int i = 0; i < 3; i++) {
    if (t < F) {
      float acc = b[i * F + t];
      for (int k = 0; k < F; k++) acc = fmaf(h[k], W[(i * F + k) * F + t], acc);
      h2[t] = silu_f(acc);
    }
    __syncthreads();
    if (t < F) h[t] = h2[t];
    __syncthreads();
  }
  if (t < N_ACTIONS) {
    float acc = bout[t];
    for (int k = 0; k < F; k++) acc = fmaf(h[k], Wout[k * N_ACTIONS + t], acc);
    out[g * N_ACTIONS + t] = acc;
  }
}

// -------------------------------------------------------------------------
extern "C" void kernel_launch(void* const* d_in, const int* in_sizes, int n_in,
                              void* d_out, int out_size, void* d_ws, size_t ws_size,
                              hipStream_t stream) {
  (void)in_sizes; (void)n_in; (void)out_size; (void)ws_size;

  // ---- inputs (setup_inputs dict order; params flattened in insertion order)
  const int*   Z      = (const int*)d_in[0];
  const int*   agc    = (const int*)d_in[1];   // atom_graph_center
  const int*   agn    = (const int*)d_in[2];   // atom_graph_nbr
  const int*   d2u    = (const int*)d_in[3];   // directed2undirected
  const int*   bga    = (const int*)d_in[4];   // bond_graph_atom
  const int*   bgi    = (const int*)d_in[5];   // bond_graph_i
  const int*   bgj    = (const int*)d_in[6];   // bond_graph_j
  const int*   owners = (const int*)d_in[7];
  const float* bb_ag  = (const float*)d_in[8];
  const float* bb_bg  = (const float*)d_in[9];
  const float* abas   = (const float*)d_in[10];
  const float* p_atom_emb    = (const float*)d_in[11];
  const float* p_bond_emb_W  = (const float*)d_in[12];
  const float* p_bond_emb_b  = (const float*)d_in[13];
  const float* p_angle_emb_W = (const float*)d_in[14];
  const float* p_angle_emb_b = (const float*)d_in[15];
  const float* p_bw_ag_W     = (const float*)d_in[16];
  const float* p_bw_ag_b     = (const float*)d_in[17];
  const float* p_bw_bg_W     = (const float*)d_in[18];
  const float* p_bw_bg_b     = (const float*)d_in[19];
  const float* p_atom_Wc     = (const float*)d_in[20];
  const float* p_atom_bc     = (const float*)d_in[21];
  const float* p_atom_Wg     = (const float*)d_in[22];
  const float* p_atom_bg     = (const float*)d_in[23];
  const float* p_bond_Wc     = (const float*)d_in[24];
  const float* p_bond_bc     = (const float*)d_in[25];
  const float* p_bond_Wg     = (const float*)d_in[26];
  const float* p_bond_bg     = (const float*)d_in[27];
  const float* p_ang_Wc      = (const float*)d_in[28];
  const float* p_ang_bc      = (const float*)d_in[29];
  const float* p_ang_Wg      = (const float*)d_in[30];
  const float* p_ang_bg      = (const float*)d_in[31];
  const float* p_ln_g        = (const float*)d_in[32];
  const float* p_ln_b        = (const float*)d_in[33];
  const float* p_head_W      = (const float*)d_in[34];
  const float* p_head_b      = (const float*)d_in[35];
  const float* p_head_Wout   = (const float*)d_in[36];
  const float* p_head_bout   = (const float*)d_in[37];

  // ---- workspace carve
  char* ws = (char*)d_ws;
  size_t off = 0;
  auto carve = [&](size_t bytes) -> void* {
    void* p = ws + off;
    off = (off + bytes + 255) & ~(size_t)255;
    return p;
  };
  const size_t ATOM_B = (size_t)N_ATOMS * F * sizeof(float);
  const size_t BOND_B = (size_t)N_UB * F * sizeof(float);
  const size_t ANG_B  = (size_t)N_ANG * F * sizeof(float);

  float* atomA = (float*)carve(ATOM_B);
  float* atomB = (float*)carve(ATOM_B);
  float* bondA = (float*)carve(BOND_B);
  float* bondB = (float*)carve(BOND_B);
  float* angle = (float*)carve(ANG_B);
  float* wag   = (float*)carve(BOND_B);
  float* wbg   = (float*)carve(BOND_B);
  const int ATOM_PK = (192 / 32) * 1024;       // dwords per atom layer = 6144
  const int BA_PK   = (256 / 32) * 1024;       // dwords per bond/ang layer = 8192
  uint32_t* atomWcP = (uint32_t*)carve((size_t)N_CONV * ATOM_PK * 4);
  uint32_t* atomWgP = (uint32_t*)carve((size_t)N_CONV * ATOM_PK * 4);
  uint32_t* bondWcP = (uint32_t*)carve((size_t)(N_CONV - 1) * BA_PK * 4);
  uint32_t* bondWgP = (uint32_t*)carve((size_t)(N_CONV - 1) * BA_PK * 4);
  uint32_t* angWcP  = (uint32_t*)carve((size_t)(N_CONV - 1) * BA_PK * 4);
  uint32_t* angWgP  = (uint32_t*)carve((size_t)(N_CONV - 1) * BA_PK * 4);
  float* sums   = (float*)carve((size_t)N_GRAPHS * F * sizeof(float));
  float* counts = (float*)carve((size_t)N_GRAPHS * sizeof(float));

  // ---- pack weights to fragment-ordered bf16 (layer K % 32 == 0 -> stacks OK)
  {
    int gA = (N_CONV * ATOM_PK + 255) / 256;        // K_total = 4*192 = 768
    pack_frag_kernel<<<gA, 256, 0, stream>>>(p_atom_Wc, atomWcP, N_CONV * 192);
    pack_frag_kernel<<<gA, 256, 0, stream>>>(p_atom_Wg, atomWgP, N_CONV * 192);
    int gB = ((N_CONV - 1) * BA_PK + 255) / 256;    // K_total = 3*256 = 768
    pack_frag_kernel<<<gB, 256, 0, stream>>>(p_bond_Wc, bondWcP, (N_CONV - 1) * 256);
    pack_frag_kernel<<<gB, 256, 0, stream>>>(p_bond_Wg, bondWgP, (N_CONV - 1) * 256);
    pack_frag_kernel<<<gB, 256, 0, stream>>>(p_ang_Wc,  angWcP,  (N_CONV - 1) * 256);
    pack_frag_kernel<<<gB, 256, 0, stream>>>(p_ang_Wg,  angWgP,  (N_CONV - 1) * 256);
  }

  // ---- embeddings
  atom_embed_kernel<<<(N_ATOMS * F + 255) / 256, 256, 0, stream>>>(Z, p_atom_emb, atomA, N_ATOMS);
  linear_small_kernel<<<(N_UB * F + 255) / 256, 256, 0, stream>>>(bb_ag, p_bond_emb_W, p_bond_emb_b, bondA, N_UB, N_RBF);
  linear_small_kernel<<<(N_ANG * F + 255) / 256, 256, 0, stream>>>(abas, p_angle_emb_W, p_angle_emb_b, angle, N_ANG, N_ABASIS);
  linear_small_kernel<<<(N_UB * F + 255) / 256, 256, 0, stream>>>(bb_ag, p_bw_ag_W, p_bw_ag_b, wag, N_UB, N_RBF);
  linear_small_kernel<<<(N_UB * F + 255) / 256, 256, 0, stream>>>(bb_bg, p_bw_bg_W, p_bw_bg_b, wbg, N_UB, N_RBF);

  float* atomIn = atomA; float* atomOut = atomB;
  float* bondIn = bondA; float* bondOut = bondB;
  const int gridAC = (N_DB + 63) / 64;    // 3125
  const int gridBG = (N_ANG + 63) / 64;   // 1563

  auto atom_conv = [&](int layer) {
    hipMemcpyAsync(atomOut, atomIn, ATOM_B, hipMemcpyDeviceToDevice, stream);
    gated_msg_kernel<3, true, true><<<gridAC, 128, 0, stream>>>(
        N_DB,
        atomIn, agc, atomIn, agn, bondIn, d2u, (const float*)nullptr, (const int*)nullptr,
        atomWcP + (size_t)layer * ATOM_PK, p_atom_bc + layer * F,
        atomWgP + (size_t)layer * ATOM_PK, p_atom_bg + layer * F,
        wag, d2u, atomOut, agc);
    float* t = atomIn; atomIn = atomOut; atomOut = t;
  };

  for (int idx = 0; idx < N_CONV - 1; idx++) {
    atom_conv(idx);

    // bond update: x = [atom[bga], bond[bgi], bond[bgj], angle]; scatter into bgi
    hipMemcpyAsync(bondOut, bondIn, BOND_B, hipMemcpyDeviceToDevice, stream);
    gated_msg_kernel<4, true, true><<<gridBG, 128, 0, stream>>>(
        N_ANG,
        atomIn, bga, bondIn, bgi, bondIn, bgj, angle, (const int*)nullptr,
        bondWcP + (size_t)idx * BA_PK, p_bond_bc + idx * F,
        bondWgP + (size_t)idx * BA_PK, p_bond_bg + idx * F,
        wbg, bgi, bondOut, bgi);
    { float* t = bondIn; bondIn = bondOut; bondOut = t; }

    // angle update (in place; identity row is the only self-reference)
    gated_msg_kernel<4, false, false><<<gridBG, 128, 0, stream>>>(
        N_ANG,
        atomIn, bga, bondIn, bgi, bondIn, bgj, angle, (const int*)nullptr,
        angWcP + (size_t)idx * BA_PK, p_ang_bc + idx * F,
        angWgP + (size_t)idx * BA_PK, p_ang_bg + idx * F,
        (const float*)nullptr, (const int*)nullptr, angle, (const int*)nullptr);

    atom_conv(N_CONV - 1);
    layernorm_kernel<<<(N_ATOMS + 3) / 4, 128, 0, stream>>>(atomIn, p_ln_g, p_ln_b, N_ATOMS);
  }

  // ---- pooling + head
  pool_zero_kernel<<<8, 256, 0, stream>>>(sums, counts);
  pool_accum_kernel<<<(N_ATOMS * F + 255) / 256, 256, 0, stream>>>(atomIn, owners, sums, counts, N_ATOMS);
  head_kernel<<<N_GRAPHS, 96, 0, stream>>>(sums, counts, p_head_W, p_head_b,
                                           p_head_Wout, p_head_bout, (float*)d_out);
}
